// Block_6975026889258
// MI455X (gfx1250) — matrix operand order repaired and checked by reading, not compiled, not run on hardware
//
#include <hip/hip_runtime.h>
#include <math.h>

// ---------------------------------------------------------------------------
// MI455X (gfx1250) two-stage kNN-graph GCN.
//  * all O(N^2 d) matmuls via v_wmma_f32_16x16x32_bf16 (bf16 in, f32 accum)
//  * adj (256 MB f32) streamed from HBM exactly once per stage (wave owns a
//    16x256 output strip); B operand staged to LDS per workgroup with
//    global_load_async_to_lds_b128 double buffering (ASYNCcnt)
//  * distance matrix never materialized: WMMA tile -> LDS -> per-row 16-heap,
//    64-row Z panels shared by 4 waves through LDS
//  * sparse symmetrized/normalized kNN graph applied as atomic scatter epilogue
// ---------------------------------------------------------------------------

typedef __bf16 bf16_t;
typedef bf16_t v16bf __attribute__((ext_vector_type(16)));
typedef bf16_t v8bf  __attribute__((ext_vector_type(8)));
typedef float  v8f   __attribute__((ext_vector_type(8)));
typedef float  v4f   __attribute__((ext_vector_type(4)));
typedef int    v4i_g __attribute__((vector_size(4 * sizeof(int))));

#define NROWS 8192
#define DIN   512
#define DOUT  256
#define KNN   16
#define ALPHA_C 0.5f
#define BETA_C  1.0f

#if __has_builtin(__builtin_amdgcn_global_load_async_to_lds_b128) && \
    __has_builtin(__builtin_amdgcn_s_wait_asynccnt)
#define USE_ASYNC_LDS 1
#else
#define USE_ASYNC_LDS 0
#endif

__device__ __forceinline__ void ldsCopyB128(const bf16_t* g, bf16_t* l) {
#if USE_ASYNC_LDS
  __builtin_amdgcn_global_load_async_to_lds_b128(
      (__attribute__((address_space(1))) v4i_g*)g,
      (__attribute__((address_space(3))) v4i_g*)l, 0, 0);
#else
  *(v8bf*)l = *(const v8bf*)g;
#endif
}
__device__ __forceinline__ void asyncWaitAll() {
#if USE_ASYNC_LDS
  __builtin_amdgcn_s_wait_asynccnt(0);
#endif
}

__device__ __forceinline__ void atomicAddF(float* p, float v) {
  __hip_atomic_fetch_add(p, v, __ATOMIC_RELAXED, __HIP_MEMORY_SCOPE_AGENT);
}

// ---- WMMA fragment loaders (layouts per CDNA5 ISA §7.12.2) ------
// A 16x32 bf16: lane L -> row (row0+L%16); V0..3: k0+8*(L/16)+0..7,
//                                          V4..7: k0+16+8*(L/16)+0..7
__device__ __forceinline__ v16bf fragA_bf16(const bf16_t* base, int ld,
                                            int row0, int k0, int lane) {
  const int m = lane & 15, h = lane >> 4;
  const bf16_t* p = base + (size_t)(row0 + m) * ld + k0 + h * 8;
  v8bf lo = *(const v8bf*)p;
  v8bf hi = *(const v8bf*)(p + 16);
  return __builtin_shufflevector(lo, hi, 0,1,2,3,4,5,6,7,8,9,10,11,12,13,14,15);
}
// same fragment converting a row-major f32 source on the fly (adj, x, H)
__device__ __forceinline__ v16bf fragA_f32(const float* base, int ld,
                                           int row0, int k0, int lane) {
  const int m = lane & 15, h = lane >> 4;
  const float* p = base + (size_t)(row0 + m) * ld + k0 + h * 8;
  v4f a = *(const v4f*)p,        b = *(const v4f*)(p + 4);
  v4f c = *(const v4f*)(p + 16), d = *(const v4f*)(p + 20);
  v16bf r;
#pragma unroll
  for (int t = 0; t < 4; ++t) {
    r[t]      = (bf16_t)a[t];
    r[4 + t]  = (bf16_t)b[t];
    r[8 + t]  = (bf16_t)c[t];
    r[12 + t] = (bf16_t)d[t];
  }
  return r;
}
// B 32x16 bf16 (NT): lane L -> row (n0+L%16), K = k0+8*(L/16)+0..7, k0+16+8*(L/16)+0..7
__device__ __forceinline__ v16bf fragB_bf16(const bf16_t* base, int ld,
                                            int n0, int k0, int lane) {
  const int n = lane & 15, h = lane >> 4;
  const bf16_t* p = base + (size_t)(n0 + n) * ld + k0 + h * 8;
  v8bf lo = *(const v8bf*)p;
  v8bf hi = *(const v8bf*)(p + 16);
  return __builtin_shufflevector(lo, hi, 0,1,2,3,4,5,6,7,8,9,10,11,12,13,14,15);
}

// ---- generic NT GEMM (Gram / whitening): 16x16 tile per wave ---------------
template <int AF32, int WF, int WH>
__global__ void __launch_bounds__(128)
gemm_nt(const void* __restrict__ Ap, int lda,
        const bf16_t* __restrict__ B, int ldb,
        float* __restrict__ Cf, int ldcf,
        bf16_t* __restrict__ Ch, int ldch,
        int Kk, float alpha) {
  const int lane = threadIdx.x & 31;
  const int wave = threadIdx.x >> 5;
  const int i0 = (blockIdx.y * 4 + wave) * 16;
  const int j0 = blockIdx.x * 16;
  v8f acc = {};
  for (int k0 = 0; k0 < Kk; k0 += 32) {
    v16bf a;
    if (AF32) a = fragA_f32((const float*)Ap, lda, i0, k0, lane);
    else      a = fragA_bf16((const bf16_t*)Ap, lda, i0, k0, lane);
    v16bf b = fragB_bf16(B, ldb, j0, k0, lane);
    acc = __builtin_amdgcn_wmma_f32_16x16x32_bf16(false, a, false, b,
                                                  (short)0, acc, false, false);
  }
  const int mh = (lane >> 4) * 8, n = lane & 15;
#pragma unroll
  for (int r = 0; r < 8; ++r) {
    const int row = i0 + mh + r, col = j0 + n;
    const float v = alpha * acc[r];
    if (WF) Cf[(size_t)row * ldcf + col] = v;
    if (WH) Ch[(size_t)row * ldch + col] = (bf16_t)v;
  }
}

// ---- wide NT GEMM: C[M x 256] = alpha * A[f32, M x K] @ B[256 x K]^T -------
// Wave owns 16 rows x all 256 cols (16 accumulators) -> A (the HBM stream,
// e.g. adj) is read exactly once. B chunk (256x32 bf16 = 16 KB) is staged to
// LDS per workgroup, double-buffered with async-to-LDS loads.
template <int WT>
__global__ void __launch_bounds__(128)
gemm_wide(const float* __restrict__ A, int lda,
          const bf16_t* __restrict__ B,      // [256 x Kk], ld = Kk
          float* __restrict__ Cf,            // [M x 256]
          bf16_t* __restrict__ Ct, int ldct, // optional transposed bf16 out
          int Kk, float alpha) {
  __shared__ alignas(16) bf16_t Bs[2][256 * 32];
  const int tid = threadIdx.x;
  const int lane = tid & 31;
  const int wave = tid >> 5;
  const int i0 = (blockIdx.x * 4 + wave) * 16;

  v8f acc[16];
#pragma unroll
  for (int j = 0; j < 16; ++j) acc[j] = (v8f){};

  const int nch = Kk / 32;
  // stage one 256x32 B chunk: 1024 b128 transfers / 128 threads = 8 each
  auto stage = [&](int buf, int k0) {
#pragma unroll
    for (int q = 0; q < 8; ++q) {
      const int e = tid + q * 128;
      const int row = e >> 2, seg = e & 3;
      ldsCopyB128(B + (size_t)row * Kk + k0 + seg * 8,
                  &Bs[buf][row * 32 + seg * 8]);
    }
  };
  stage(0, 0);
  asyncWaitAll();
  __syncthreads();

  for (int c = 0; c < nch; ++c) {
    const int k0 = c * 32;
    if (c + 1 < nch) stage((c + 1) & 1, k0 + 32);
    if (k0 + 256 < Kk)  // prefetch the A (HBM) stream 8 chunks ahead
      __builtin_prefetch(A + (size_t)(i0 + (lane & 15)) * lda + k0 + 256, 0, 1);
    v16bf a = fragA_f32(A, lda, i0, k0, lane);
    const bf16_t* bb = &Bs[c & 1][0];
#pragma unroll
    for (int j = 0; j < 16; ++j) {
      v16bf b = fragB_bf16(bb, 32, j * 16, 0, lane);
      acc[j] = __builtin_amdgcn_wmma_f32_16x16x32_bf16(false, a, false, b,
                                                       (short)0, acc[j],
                                                       false, false);
    }
    asyncWaitAll();      // staged chunk c+1 complete
    __syncthreads();     // all waves done reading buf c before re-staging it
  }

  const int mh = (lane >> 4) * 8, n = lane & 15;
#pragma unroll
  for (int j = 0; j < 16; ++j) {
#pragma unroll
    for (int r = 0; r < 8; ++r) {
      const int row = i0 + mh + r, col = j * 16 + n;
      const float v = alpha * acc[j][r];
      Cf[(size_t)row * DOUT + col] = v;
      if (WT) Ct[(size_t)col * ldct + row] = (bf16_t)v;
    }
  }
}

// ---- fused Z*Z^T + per-row top-16 (smallest d2) ---------------------------
// 4 waves/block; each wave owns 16 query rows (A fragments VGPR-resident for
// the whole sweep); 64-row Z panels shared by the block through LDS.
template <int KD>
__global__ void __launch_bounds__(128)
disttopk(const bf16_t* __restrict__ Z, const float* __restrict__ sq,
         float* __restrict__ outD2, int* __restrict__ outIdx,
         float* __restrict__ sigsum, int nrows) {
  __shared__ alignas(16) bf16_t pan[64 * KD];
  __shared__ float gt[4][16][17];
  __shared__ float hv[4][16][16];
  __shared__ int   hx[4][16][16];
  const int tid = threadIdx.x;
  const int lane = tid & 31, wave = tid >> 5;
  const int i0 = (blockIdx.x * 4 + wave) * 16;

  if (lane < 16) {
#pragma unroll
    for (int t = 0; t < 16; ++t) { hv[wave][lane][t] = 3.0e38f; hx[wave][lane][t] = 0; }
  }
  v16bf afr[KD / 32];
#pragma unroll
  for (int c = 0; c < KD / 32; ++c) afr[c] = fragA_bf16(Z, KD, i0, c * 32, lane);
  const float sqi = (lane < 16) ? sq[i0 + lane] : 0.f;

  for (int jp = 0; jp < nrows; jp += 64) {
    // stage 64 x KD bf16 panel (64*KD/8 b128 transfers across 128 threads)
    for (int e = tid; e < 64 * KD / 8; e += 128) {
      const int row = e / (KD / 8), seg = e % (KD / 8);
      ldsCopyB128(Z + (size_t)(jp + row) * KD + seg * 8, &pan[row * KD + seg * 8]);
    }
    asyncWaitAll();
    __syncthreads();
#pragma unroll
    for (int jt = 0; jt < 4; ++jt) {
      const int j0 = jp + jt * 16;
      v8f acc = {};
#pragma unroll
      for (int c = 0; c < KD / 32; ++c) {
        v16bf b = fragB_bf16(pan, KD, jt * 16, c * 32, lane);
        acc = __builtin_amdgcn_wmma_f32_16x16x32_bf16(false, afr[c], false, b,
                                                      (short)0, acc, false, false);
      }
      const int mh = (lane >> 4) * 8, n = lane & 15;
#pragma unroll
      for (int r = 0; r < 8; ++r) gt[wave][mh + r][n] = acc[r];
      __syncthreads();
      if (lane < 16) {
        float* hvr = hv[wave][lane];
        int*   hir = hx[wave][lane];
        for (int t = 0; t < 16; ++t) {
          const float d2 = sqi + sq[j0 + t] - 2.f * gt[wave][lane][t];
          if (d2 < hvr[0]) {            // replace max-heap root, sift down
            int p = 0;
            for (;;) {
              int c2 = 2 * p + 1;
              if (c2 >= 16) break;
              if (c2 + 1 < 16 && hvr[c2 + 1] > hvr[c2]) ++c2;
              if (hvr[c2] <= d2) break;
              hvr[p] = hvr[c2]; hir[p] = hir[c2]; p = c2;
            }
            hvr[p] = d2; hir[p] = j0 + t;
          }
        }
      }
      __syncthreads();
    }
  }
  if (lane < 16) {
    float s = 0.f;
#pragma unroll
    for (int t = 0; t < 16; ++t) {
      float d2 = hv[wave][lane][t];
      d2 = d2 > 0.f ? d2 : 0.f;         // clip (matches jnp.clip)
      outD2[(size_t)(i0 + lane) * KNN + t] = d2;
      outIdx[(size_t)(i0 + lane) * KNN + t] = hx[wave][lane][t];
      s += sqrtf(d2);
    }
    atomicAddF(sigsum, s);
  }
}

// ---- small helper kernels -------------------------------------------------
__global__ void transpose_cvt(const float* __restrict__ src, bf16_t* __restrict__ dst,
                              int R, int C) {  // dst[c*R+r] = src[r*C+c]
  size_t g = (size_t)blockIdx.x * blockDim.x + threadIdx.x;
  if (g >= (size_t)R * C) return;
  int r = (int)(g / C), c = (int)(g % C);
  dst[(size_t)c * R + r] = (bf16_t)src[g];
}
__global__ void cvt_bf16(const float* __restrict__ s, bf16_t* __restrict__ d, int n) {
  int g = blockIdx.x * blockDim.x + threadIdx.x;
  if (g < n) d[g] = (bf16_t)s[g];
}
__global__ void zero_f32(float* __restrict__ p, int n) {
  int g = blockIdx.x * blockDim.x + threadIdx.x;
  if (g < n) p[g] = 0.f;
}
__global__ void add_diag(float* __restrict__ A, int d, float beta) {
  int g = blockIdx.x * blockDim.x + threadIdx.x;
  if (g < d) A[(size_t)g * d + g] += beta;
}
__global__ void w2sym_cvt(const float* __restrict__ w, bf16_t* __restrict__ o, int d) {
  int g = blockIdx.x * blockDim.x + threadIdx.x;
  if (g >= d * d) return;
  int a = g / d, b = g % d;
  o[g] = (bf16_t)(0.5f * (w[(size_t)a * d + b] + w[(size_t)b * d + a]));
}
// single-block right-looking Cholesky; 32 waves chew the trailing update
__global__ void cholesky_k(float* __restrict__ A, int d) {
  for (int j = 0; j < d; ++j) {
    if (threadIdx.x == 0) A[(size_t)j * d + j] = sqrtf(A[(size_t)j * d + j]);
    __syncthreads();
    const float djj = A[(size_t)j * d + j];
    for (int i = j + 1 + threadIdx.x; i < d; i += blockDim.x)
      A[(size_t)i * d + j] /= djj;
    __syncthreads();
    for (int i = j + 1 + threadIdx.x; i < d; i += blockDim.x) {
      const float lij = A[(size_t)i * d + j];
      for (int k2 = j + 1; k2 <= i; ++k2)
        A[(size_t)i * d + k2] -= lij * A[(size_t)k2 * d + j];
    }
    __syncthreads();
  }
}
// explicit lower-triangular inverse, one column per thread
__global__ void trinv_k(const float* __restrict__ L, float* __restrict__ Li, int d) {
  int c = blockIdx.x * blockDim.x + threadIdx.x;
  if (c >= d) return;
  for (int i = 0; i < c; ++i) Li[(size_t)i * d + c] = 0.f;
  for (int i = c; i < d; ++i) {
    float s = (i == c) ? 1.f : 0.f;
    for (int j = c; j < i; ++j) s -= L[(size_t)i * d + j] * Li[(size_t)j * d + c];
    Li[(size_t)i * d + c] = s / L[(size_t)i * d + i];
  }
}
__global__ void __launch_bounds__(32)
sqnorm_k(const bf16_t* __restrict__ Z, float* __restrict__ sq, int d) {
  const int row = blockIdx.x, lane = threadIdx.x;
  float s = 0.f;
  for (int k = lane; k < d; k += 32) {
    float z = (float)Z[(size_t)row * d + k];
    s += z * z;
  }
  for (int off = 16; off; off >>= 1) s += __shfl_xor(s, off, 32);
  if (lane == 0) sq[row] = s;
}
// d2 -> Gaussian kernel in-place, plus degree accumulation of 0.5*(S + S^T)
__global__ void kern_deg_k(float* __restrict__ kd, const int* __restrict__ idx,
                           const float* __restrict__ sigsum, float* __restrict__ deg,
                           int nk) {
  int g = blockIdx.x * blockDim.x + threadIdx.x;
  if (g >= nk) return;
  const float sigma = *sigsum / (float)nk;
  const float inv = 1.f / (2.f * sigma * sigma);
  const float kv = expf(-kd[g] * inv);
  kd[g] = kv;
  const int r = g >> 4, c = idx[g];
  atomicAddF(&deg[r], 0.5f * kv);
  atomicAddF(&deg[c], 0.5f * kv);
}
__global__ void dis_k(const float* __restrict__ deg, float* __restrict__ dis, int n) {
  int g = blockIdx.x * blockDim.x + threadIdx.x;
  if (g < n) { float d = deg[g]; dis[g] = d > 0.f ? rsqrtf(d) : 0.f; }
}
// sparse normalized-graph SpMM epilogue: ACC += 0.5*D(S+S^T)D @ P (atomics)
__global__ void __launch_bounds__(256)
scatter_k(const float* __restrict__ kv, const int* __restrict__ idx,
          const float* __restrict__ dis, const float* __restrict__ P,
          float* __restrict__ ACC, int cols) {
  const int e = blockIdx.x;
  const int r = e >> 4, c = idx[e];
  const float w = 0.5f * kv[e] * dis[r] * dis[c];
  const float* Pr = P + (size_t)r * cols;
  const float* Pc = P + (size_t)c * cols;
  float* Ar = ACC + (size_t)r * cols;
  float* Ac = ACC + (size_t)c * cols;
  for (int j = threadIdx.x; j < cols; j += blockDim.x) {
    atomicAddF(&Ac[j], w * Pr[j]);   // S^T direction
    atomicAddF(&Ar[j], w * Pc[j]);   // S   direction
  }
}
__global__ void tanh_k(const float* __restrict__ a, float* __restrict__ o, int n) {
  int g = blockIdx.x * blockDim.x + threadIdx.x;
  if (g < n) o[g] = tanhf(a[g]);
}

// ---------------------------------------------------------------------------
extern "C" void kernel_launch(void* const* d_in, const int* in_sizes, int n_in,
                              void* d_out, int out_size, void* d_ws, size_t ws_size,
                              hipStream_t stream) {
  const float* x   = (const float*)d_in[0];   // [8192,512]
  const float* adj = (const float*)d_in[1];   // [8192,8192]
  const float* w1  = (const float*)d_in[2];   // [512,256]
  const float* w2  = (const float*)d_in[3];   // [256,256]
  float* outp = (float*)d_out;                // [8192,256]

  char* ws = (char*)d_ws;
  size_t off = 0;
  auto take = [&](size_t bytes) -> char* {
    char* p = ws + off;
    off = (off + bytes + 255) & ~(size_t)255;
    return p;
  };
  bf16_t* Zh    = (bf16_t*)take((size_t)NROWS * DIN * 2);   // whitened feats
  bf16_t* Th    = (bf16_t*)take((size_t)DIN * NROWS * 2);   // transposed feats
  bf16_t* Pt    = (bf16_t*)take((size_t)DOUT * NROWS * 2);  // (X W)^T bf16
  float*  Pf    = (float*) take((size_t)NROWS * DOUT * 4);  // X W f32
  float*  ACC   = (float*) take((size_t)NROWS * DOUT * 4);  // pre-activation
  float*  H1f   = (float*) take((size_t)NROWS * DOUT * 4);  // stage-1 output
  float*  Amat  = (float*) take((size_t)DIN * DIN * 4);     // Gram / Cholesky
  float*  Linv  = (float*) take((size_t)DIN * DIN * 4);
  bf16_t* Linvh = (bf16_t*)take((size_t)DIN * DIN * 2);
  bf16_t* Wt    = (bf16_t*)take((size_t)DOUT * DIN * 2);    // W1^T / W2sym
  float*  sqn   = (float*) take((size_t)NROWS * 4);
  float*  tkD   = (float*) take((size_t)NROWS * KNN * 4);
  int*    tkI   = (int*)   take((size_t)NROWS * KNN * 4);
  float*  deg   = (float*) take((size_t)NROWS * 4);
  float*  dis   = (float*) take((size_t)NROWS * 4);
  float*  sig   = (float*) take(256);

  const int NK = NROWS * KNN;
  auto g1 = [](int n) { return dim3((n + 255) / 256); };

  // ============================ stage 1 (d = 512) ==========================
  transpose_cvt<<<g1(NROWS * DIN), 256, 0, stream>>>(x, Th, NROWS, DIN);
  transpose_cvt<<<g1(DIN * DOUT), 256, 0, stream>>>(w1, Wt, DIN, DOUT);
  // Gram: A = X^T X  (512x512, K=8192)
  gemm_nt<0, 1, 0><<<dim3(DIN / 16, DIN / 64), 128, 0, stream>>>(
      Th, NROWS, Th, NROWS, Amat, DIN, nullptr, 0, NROWS, 1.f);
  add_diag<<<g1(DIN), 256, 0, stream>>>(Amat, DIN, BETA_C);
  cholesky_k<<<1, 1024, 0, stream>>>(Amat, DIN);
  trinv_k<<<dim3(2), 256, 0, stream>>>(Amat, Linv, DIN);
  cvt_bf16<<<g1(DIN * DIN), 256, 0, stream>>>(Linv, Linvh, DIN * DIN);
  // Z = X @ Linv^T  -> bf16
  gemm_nt<1, 0, 1><<<dim3(DIN / 16, NROWS / 64), 128, 0, stream>>>(
      x, DIN, Linvh, DIN, nullptr, 0, Zh, DIN, DIN, 1.f);
  sqnorm_k<<<dim3(NROWS), 32, 0, stream>>>(Zh, sqn, DIN);
  zero_f32<<<1, 1, 0, stream>>>(sig, 1);
  zero_f32<<<g1(NROWS), 256, 0, stream>>>(deg, NROWS);
  disttopk<DIN><<<dim3(NROWS / 64), 128, 0, stream>>>(Zh, sqn, tkD, tkI, sig, NROWS);
  kern_deg_k<<<g1(NK), 256, 0, stream>>>(tkD, tkI, sig, deg, NK);
  dis_k<<<g1(NROWS), 256, 0, stream>>>(deg, dis, NROWS);
  // P = X W1 (f32 + transposed bf16)
  gemm_wide<1><<<dim3(NROWS / 64), 128, 0, stream>>>(
      x, DIN, Wt, Pf, Pt, NROWS, DIN, 1.f);
  // ACC = 0.5 * adj @ P   (adj streamed from HBM once)
  gemm_wide<0><<<dim3(NROWS / 64), 128, 0, stream>>>(
      adj, NROWS, Pt, ACC, nullptr, 0, NROWS, ALPHA_C);
  scatter_k<<<dim3(NK), 256, 0, stream>>>(tkD, tkI, dis, Pf, ACC, DOUT);
  tanh_k<<<g1(NROWS * DOUT), 256, 0, stream>>>(ACC, H1f, NROWS * DOUT);

  // ============================ stage 2 (d = 256) ==========================
  transpose_cvt<<<g1(NROWS * DOUT), 256, 0, stream>>>(H1f, Th, NROWS, DOUT);
  w2sym_cvt<<<g1(DOUT * DOUT), 256, 0, stream>>>(w2, Wt, DOUT);
  gemm_nt<0, 1, 0><<<dim3(DOUT / 16, DOUT / 64), 128, 0, stream>>>(
      Th, NROWS, Th, NROWS, Amat, DOUT, nullptr, 0, NROWS, 1.f);
  add_diag<<<g1(DOUT), 256, 0, stream>>>(Amat, DOUT, BETA_C);
  cholesky_k<<<1, 1024, 0, stream>>>(Amat, DOUT);
  trinv_k<<<dim3(1), 256, 0, stream>>>(Amat, Linv, DOUT);
  cvt_bf16<<<g1(DOUT * DOUT), 256, 0, stream>>>(Linv, Linvh, DOUT * DOUT);
  gemm_nt<1, 0, 1><<<dim3(DOUT / 16, NROWS / 64), 128, 0, stream>>>(
      H1f, DOUT, Linvh, DOUT, nullptr, 0, Zh, DOUT, DOUT, 1.f);
  sqnorm_k<<<dim3(NROWS), 32, 0, stream>>>(Zh, sqn, DOUT);
  zero_f32<<<1, 1, 0, stream>>>(sig, 1);
  zero_f32<<<g1(NROWS), 256, 0, stream>>>(deg, NROWS);
  disttopk<DOUT><<<dim3(NROWS / 64), 128, 0, stream>>>(Zh, sqn, tkD, tkI, sig, NROWS);
  kern_deg_k<<<g1(NK), 256, 0, stream>>>(tkD, tkI, sig, deg, NK);
  dis_k<<<g1(NROWS), 256, 0, stream>>>(deg, dis, NROWS);
  gemm_wide<1><<<dim3(NROWS / 64), 128, 0, stream>>>(
      H1f, DOUT, Wt, Pf, Pt, NROWS, DOUT, 1.f);
  gemm_wide<0><<<dim3(NROWS / 64), 128, 0, stream>>>(
      adj, NROWS, Pt, ACC, nullptr, 0, NROWS, ALPHA_C);
  scatter_k<<<dim3(NK), 256, 0, stream>>>(tkD, tkI, dis, Pf, ACC, DOUT);
  tanh_k<<<g1(NROWS * DOUT), 256, 0, stream>>>(ACC, outp, NROWS * DOUT);
}
